// GlobalLocalSelfAttention_2740189134927
// MI455X (gfx1250) — matrix operand order repaired and checked by reading, not compiled, not run on hardware
//
#include <hip/hip_runtime.h>

#define DEV static __device__ __forceinline__

typedef _Float16 h16v __attribute__((ext_vector_type(16)));
typedef _Float16 h8v  __attribute__((ext_vector_type(8)));
typedef float    f8v  __attribute__((ext_vector_type(8)));
typedef float    f4v  __attribute__((ext_vector_type(4)));

// problem constants
constexpr int Bc = 4, Tc = 4096, Dc = 1024, Gc = 128, Kc = 64, Hc = 16, Fc = 64;
constexpr int Pc = Tc + Kc;     // 4160 padded local tokens per batch
constexpr int Nc = Pc / Kc;     // 65 local blocks
constexpr int GQ = Bc * Gc;     // 512 global-attention output rows
constexpr int SLICES = 4;       // key-splits for global attention
constexpr int CHUNKS_G = (Gc + Pc) / 32;          // 134 key chunks (global attn)
constexpr int CPS = (CHUNKS_G + SLICES - 1) / SLICES;  // 34 chunks per slice
constexpr float NEGV = -1e10f;

DEV h16v cat8(h8v a, h8v b) {
  return __builtin_shufflevector(a, b, 0,1,2,3,4,5,6,7,8,9,10,11,12,13,14,15);
}
DEV h8v load8(const float* p) {
  f4v a = *(const f4v*)p;
  f4v b = *(const f4v*)(p + 4);
  f8v f = __builtin_shufflevector(a, b, 0,1,2,3,4,5,6,7);
  return __builtin_convertvector(f, h8v);
}
DEV h8v load8(const _Float16* p) { return *(const h8v*)p; }

DEV f8v wmma16(h16v a, h16v b, f8v c) {
  return __builtin_amdgcn_wmma_f32_16x16x32_f16(false, a, false, b, (short)0, c,
                                                false, false);
}

// ---------------------------------------------------------------------------
// Generic WMMA GEMM with bias:  C[M,1024] = cscale * (A[M,1024] @ W + bias)
// Block = 256 threads (8 waves); block tile 256(M) x 64(N); wave tile 32 x 64
// (2 A fragments -> 8 back-to-back WMMAs per 32-deep k-step).
// Row mapping: out-row ro -> batch b=ro/rowsOutPerB, t=ro%rowsOutPerB;
// source row = b*rowsSrcPerB + t, zero if t >= rowsSrcPerB (padding).
// OUTMODE: 0 = f16 row-major, 1 = f16 transposed [b][h][f][tok], 2 = f32 row-major
// ---------------------------------------------------------------------------
template<typename AT, int OUTMODE>
__global__ __launch_bounds__(256) void gemm_bias_wmma(
    const AT* __restrict__ A, const float* __restrict__ W,
    const float* __restrict__ bias, void* __restrict__ Cout,
    int rowsOutPerB, int rowsSrcPerB, float cscale)
{
  __shared__ __align__(16) _Float16 wlds[64 * 32];  // [col 0..63][d 0..31]
  const int tid  = threadIdx.x;
  const int wave = tid >> 5;
  const int lane = tid & 31;
  const int ln   = lane & 15;
  const bool hiL = lane >= 16;
  const int col0 = blockIdx.y * 64;
  const int ro0  = blockIdx.x * 256 + wave * 32;

  bool valid[2];
  long srcBase[2];
#pragma unroll
  for (int rt = 0; rt < 2; ++rt) {
    const int roA = ro0 + rt * 16 + ln;
    const int bb  = roA / rowsOutPerB;
    const int tt  = roA % rowsOutPerB;
    valid[rt]   = tt < rowsSrcPerB;
    srcBase[rt] = (long)(bb * rowsSrcPerB + tt) * Dc;
  }

  f8v acc[2][4] = {};
  const int eBase = tid * 8;
  const int ccS   = eBase >> 5;      // staging column for this thread
  const int ddS   = eBase & 31;      // first staging depth for this thread

  for (int d0 = 0; d0 < Dc; d0 += 32) {
    // stage W tile transposed into LDS: wlds[cc*32+dd] = W[d0+dd][col0+cc]
#pragma unroll
    for (int i = 0; i < 8; ++i) {
      wlds[eBase + i] = (_Float16)W[(long)(d0 + ddS + i) * Dc + col0 + ccS];
    }
    if (d0 + 32 < Dc) {  // hint next W tile toward the caches (global_prefetch_b8)
      __builtin_prefetch(W + (long)(d0 + 32 + ddS) * Dc + col0 + ccS, 0, 1);
    }
    __syncthreads();

    // A fragments (contiguous k: halves {0..15} / {16..31})
    h16v af[2];
#pragma unroll
    for (int rt = 0; rt < 2; ++rt) {
      h16v z = {};
      af[rt] = z;
      if (valid[rt]) {
        int off = d0 + (hiL ? 16 : 0);
        af[rt] = cat8(load8(A + srcBase[rt] + off),
                      load8(A + srcBase[rt] + off + 8));
      }
    }
#pragma unroll
    for (int nt = 0; nt < 4; ++nt) {
      const _Float16* wp = &wlds[(nt * 16 + ln) * 32 + (hiL ? 16 : 0)];
      h16v bf = cat8(*(const h8v*)wp, *(const h8v*)(wp + 8));
      acc[0][nt] = wmma16(af[0], bf, acc[0][nt]);
      acc[1][nt] = wmma16(af[1], bf, acc[1][nt]);
    }
    __syncthreads();
  }

#pragma unroll
  for (int rt = 0; rt < 2; ++rt) {
#pragma unroll
    for (int nt = 0; nt < 4; ++nt) {
#pragma unroll
      for (int r = 0; r < 8; ++r) {
        int row = ro0 + rt * 16 + r + (hiL ? 8 : 0);
        int col = col0 + nt * 16 + ln;
        float v = (acc[rt][nt][r] + bias[col]) * cscale;
        if constexpr (OUTMODE == 2) {
          ((float*)Cout)[(long)row * Dc + col] = v;
        } else if constexpr (OUTMODE == 0) {
          ((_Float16*)Cout)[(long)row * Dc + col] = (_Float16)v;
        } else {
          int b2 = row / rowsOutPerB, t2 = row % rowsOutPerB;
          int hh = col >> 6, ff = col & 63;
          ((_Float16*)Cout)[((long)((b2 * Hc + hh) * Fc + ff)) * rowsOutPerB + t2] =
              (_Float16)v;
        }
      }
    }
  }
}

// ---------------------------------------------------------------------------
// Flash attention over [global keys ; local keys] with WMMA.
// LOCAL=true : grid = B*N*H   blocks; one 64-token block vs (128 G + 64 own).
//              Normalized f16 output.
// LOCAL=false: grid = B*2*H*SLICES; 64 global queries vs one key-slice.
//              Writes unnormalized partials (O, m, l) for the combine pass.
// Block = 4 waves, each wave owns 16 queries. 32-key chunks.
// ---------------------------------------------------------------------------
template<bool LOCAL>
__global__ __launch_bounds__(128) void attn_flash_wmma(
    const _Float16* __restrict__ qbuf,
    const _Float16* __restrict__ gkB, const _Float16* __restrict__ lkB,
    const _Float16* __restrict__ gvtB, const _Float16* __restrict__ lvtB,
    const int* __restrict__ mask, _Float16* __restrict__ outB,
    float* __restrict__ partO, float* __restrict__ partM,
    float* __restrict__ partL)
{
  __shared__ __align__(16) _Float16 plds[4][16 * 32];
  const int tid  = threadIdx.x;
  const int wave = tid >> 5;
  const int lane = tid & 31;
  const int ln   = lane & 15;
  const bool hiL = lane >= 16;

  int b, n, qRow0, slice, c0, c1;
  if constexpr (LOCAL) {
    const int bx = blockIdx.x;
    const int h_ = bx & 15;  (void)h_;
    n = (bx >> 4) % Nc;
    b = bx / (Hc * Nc);
    qRow0 = b * Pc + n * Kc + wave * 16;
    slice = 0;
    c0 = 0; c1 = (Gc + Kc) / 32;  // 6 chunks
  } else {
    const int bgh = Bc * 2 * Hc;
    slice = blockIdx.x / bgh;
    const int bx = blockIdx.x % bgh;
    int qt = (bx >> 4) & 1;
    b = bx >> 5;
    n = 0;
    qRow0 = b * Gc + qt * 64 + wave * 16;
    c0 = slice * CPS;
    c1 = min(CHUNKS_G, c0 + CPS);
  }
  const int h = (LOCAL ? blockIdx.x : (blockIdx.x % (Bc * 2 * Hc))) & 15;

  // Q fragments (A-operand), f 0..31 and 32..63 (1/sqrt(F) folded in upstream)
  const _Float16* qp = qbuf + (long)(qRow0 + ln) * Dc + h * Fc;
  const int aoff = hiL ? 16 : 0;
  h16v aq0 = cat8(load8(qp + aoff),      load8(qp + 8 + aoff));
  h16v aq1 = cat8(load8(qp + 32 + aoff), load8(qp + 40 + aoff));

  int qm[8];
#pragma unroll
  for (int r = 0; r < 8; ++r) {
    if constexpr (LOCAL) {
      int tq = n * Kc + wave * 16 + r + (hiL ? 8 : 0);
      qm[r] = (tq < Tc) ? mask[b * Tc + tq] : 0;
    } else {
      qm[r] = 1;
    }
  }

  float m[8], l[8];
#pragma unroll
  for (int r = 0; r < 8; ++r) { m[r] = -3.0e38f; l[r] = 0.0f; }
  f8v O[4] = {};

  for (int c = c0; c < c1; ++c) {
    const int j0 = c * 32;
    const bool gkeys = j0 < Gc;
    const int TB = (LOCAL ? n * Kc : 0) + (j0 - Gc);  // only used when !gkeys

    // ---- S = Q @ K^T for two 16-key sub-chunks ----
    f8v sf[2];
    int km[2];
#pragma unroll
    for (int s = 0; s < 2; ++s) {
      const _Float16* kp;
      if (gkeys) {
        int j = j0 + s * 16 + ln;
        kp = gkB + (long)(b * Gc + j) * Dc + h * Fc;
        km[s] = 1;
      } else {
        int tk = TB + s * 16 + ln;
        kp = lkB + (long)(b * Pc + tk) * Dc + h * Fc;
        km[s] = (tk < Tc) ? mask[b * Tc + tk] : 0;
      }
      int ko = hiL ? 16 : 0;
      h16v bf0 = cat8(load8(kp + ko),      load8(kp + ko + 8));
      h16v bf1 = cat8(load8(kp + 32 + ko), load8(kp + 32 + ko + 8));
      f8v s0 = {};
      s0 = wmma16(aq0, bf0, s0);
      s0 = wmma16(aq1, bf1, s0);
      sf[s] = s0;
    }

    // ---- online softmax update (per output row) ----
#pragma unroll
    for (int r = 0; r < 8; ++r) {
      float x0 = (km[0] && qm[r]) ? sf[0][r] : NEGV;
      float x1 = (km[1] && qm[r]) ? sf[1][r] : NEGV;
      float rm = fmaxf(x0, x1);
      rm = fmaxf(rm, __shfl_xor(rm, 1, 32));
      rm = fmaxf(rm, __shfl_xor(rm, 2, 32));
      rm = fmaxf(rm, __shfl_xor(rm, 4, 32));
      rm = fmaxf(rm, __shfl_xor(rm, 8, 32));
      float mn = fmaxf(m[r], rm);
      float al = __expf(m[r] - mn);
      float p0 = __expf(x0 - mn);
      float p1 = __expf(x1 - mn);
      float rs = p0 + p1;
      rs += __shfl_xor(rs, 1, 32);
      rs += __shfl_xor(rs, 2, 32);
      rs += __shfl_xor(rs, 4, 32);
      rs += __shfl_xor(rs, 8, 32);
      m[r] = mn;
      l[r] = l[r] * al + rs;
      O[0][r] *= al; O[1][r] *= al; O[2][r] *= al; O[3][r] *= al;
      int prow = r + (hiL ? 8 : 0);
      plds[wave][prow * 32 + ln]      = (_Float16)p0;  // keys j0..j0+15
      plds[wave][prow * 32 + 16 + ln] = (_Float16)p1;  // keys j0+16..j0+31
    }

    // ---- C-layout -> A-layout via wave-private LDS tile ----
    const _Float16* pp = &plds[wave][ln * 32];
    h16v pa = cat8(*(const h8v*)(pp + aoff), *(const h8v*)(pp + 8 + aoff));

    // ---- O += P @ V  (V stored transposed: contiguous key runs) ----
    int kvo = hiL ? 16 : 0;
#pragma unroll
    for (int ft = 0; ft < 4; ++ft) {
      int f = ft * 16 + ln;
      const _Float16* vp;
      if (gkeys) vp = gvtB + ((long)((b * Hc + h) * Fc + f)) * Gc + j0 + kvo;
      else       vp = lvtB + ((long)((b * Hc + h) * Fc + f)) * Pc + TB + kvo;
      h16v vb = cat8(*(const h8v*)vp, *(const h8v*)(vp + 8));
      O[ft] = wmma16(pa, vb, O[ft]);
    }
  }

  if constexpr (LOCAL) {
    // normalize and store per-head attention output (f16)
#pragma unroll
    for (int r = 0; r < 8; ++r) {
      float inv = 1.0f / l[r];
      long orow = qRow0 + r + (hiL ? 8 : 0);
#pragma unroll
      for (int ft = 0; ft < 4; ++ft) {
        outB[orow * Dc + h * Fc + ft * 16 + ln] = (_Float16)(O[ft][r] * inv);
      }
    }
  } else {
    // store unnormalized partials for the combine pass
#pragma unroll
    for (int r = 0; r < 8; ++r) {
      long row   = qRow0 + r + (hiL ? 8 : 0);
      long pbase = ((long)slice * GQ + row) * Dc + h * Fc;
#pragma unroll
      for (int ft = 0; ft < 4; ++ft) {
        partO[pbase + ft * 16 + ln] = O[ft][r];
      }
      if (ln == 0) {
        long mlIdx = ((long)slice * GQ + row) * Hc + h;
        partM[mlIdx] = m[r];
        partL[mlIdx] = l[r];
      }
    }
  }
}

// ---------------------------------------------------------------------------
// Merge the SLICES key-split partials of global attention into gattn (f16).
// grid = GQ rows, 256 threads cover 1024 columns.
// ---------------------------------------------------------------------------
__global__ __launch_bounds__(256) void combine_gattn(
    const float* __restrict__ partO, const float* __restrict__ partM,
    const float* __restrict__ partL, _Float16* __restrict__ gattn)
{
  const int row = blockIdx.x;
#pragma unroll
  for (int i = 0; i < 4; ++i) {
    const int col = threadIdx.x + i * 256;
    const int h = col >> 6;
    float ms[SLICES];
    float M = -3.0e38f;
#pragma unroll
    for (int s = 0; s < SLICES; ++s) {
      ms[s] = partM[((long)s * GQ + row) * Hc + h];
      M = fmaxf(M, ms[s]);
    }
    float L = 0.0f, acc = 0.0f;
#pragma unroll
    for (int s = 0; s < SLICES; ++s) {
      float w = __expf(ms[s] - M);
      L   += partL[((long)s * GQ + row) * Hc + h] * w;
      acc += partO[((long)s * GQ + row) * Dc + col] * w;
    }
    gattn[(long)row * Dc + col] = (_Float16)(acc / L);
  }
}

// ---------------------------------------------------------------------------
extern "C" void kernel_launch(void* const* d_in, const int* in_sizes, int n_in,
                              void* d_out, int out_size, void* d_ws, size_t ws_size,
                              hipStream_t stream)
{
  const float* gx  = (const float*)d_in[0];
  const float* x   = (const float*)d_in[1];
  const int*  mask = (const int*)d_in[2];
  const float* wq  = (const float*)d_in[3];
  const float* bq  = (const float*)d_in[4];
  const float* wk  = (const float*)d_in[5];
  const float* bk  = (const float*)d_in[6];
  const float* wv  = (const float*)d_in[7];
  const float* bv  = (const float*)d_in[8];
  const float* wo  = (const float*)d_in[9];
  const float* bo  = (const float*)d_in[10];

  const long szL = (long)Bc * Pc * Dc;  // 17,039,360 halves
  const long szG = (long)Bc * Gc * Dc;  //    524,288 halves
  _Float16* lq    = (_Float16*)d_ws;
  _Float16* lk    = lq    + szL;
  _Float16* lvt   = lk    + szL;   // V^T: [b][h][f][tok 0..P)
  _Float16* lattn = lvt   + szL;
  _Float16* gq    = lattn + szL;
  _Float16* gk    = gq    + szG;
  _Float16* gvt   = gk    + szG;   // V^T: [b][h][f][tok 0..G)
  _Float16* gattn = gvt   + szG;
  float* partO = (float*)(gattn + szG);                 // [SLICES][GQ][1024] f32
  float* partM = partO + (long)SLICES * GQ * Dc;        // [SLICES][GQ][H]
  float* partL = partM + (long)SLICES * GQ * Hc;        // [SLICES][GQ][H]

  const float qscale = 0.125f;  // 1/sqrt(F=64)
  dim3 blk(256);

  // local QKV projections: M = B*P = 16640 -> 65 row-blocks, 16 col-tiles
  gemm_bias_wmma<float, 0><<<dim3(65, 16), blk, 0, stream>>>(x, wq, bq, lq,  Pc, Tc, qscale);
  gemm_bias_wmma<float, 0><<<dim3(65, 16), blk, 0, stream>>>(x, wk, bk, lk,  Pc, Tc, 1.0f);
  gemm_bias_wmma<float, 1><<<dim3(65, 16), blk, 0, stream>>>(x, wv, bv, lvt, Pc, Tc, 1.0f);
  // global QKV projections: M = B*G = 512 -> 2 row-blocks
  gemm_bias_wmma<float, 0><<<dim3(2, 16), blk, 0, stream>>>(gx, wq, bq, gq,  Gc, Gc, qscale);
  gemm_bias_wmma<float, 0><<<dim3(2, 16), blk, 0, stream>>>(gx, wk, bk, gk,  Gc, Gc, 1.0f);
  gemm_bias_wmma<float, 1><<<dim3(2, 16), blk, 0, stream>>>(gx, wv, bv, gvt, Gc, Gc, 1.0f);

  // local attention (normalized f16 out)
  attn_flash_wmma<true ><<<dim3(Bc * Nc * Hc), dim3(128), 0, stream>>>(
      lq, gk, lk, gvt, lvt, mask, lattn, nullptr, nullptr, nullptr);
  // global attention: key-split partials + combine
  attn_flash_wmma<false><<<dim3(Bc * 2 * Hc * SLICES), dim3(128), 0, stream>>>(
      gq, gk, lk, gvt, lvt, mask, nullptr, partO, partM, partL);
  combine_gattn<<<dim3(GQ), dim3(256), 0, stream>>>(partO, partM, partL, gattn);

  // output projections -> d_out (global_out then local_out, f32)
  float* gout = (float*)d_out;              // (B,G,D)
  float* lout = gout + (long)Bc * Gc * Dc;  // (B,T,D)
  gemm_bias_wmma<_Float16, 2><<<dim3(2,  16), blk, 0, stream>>>(gattn, wo, bo, gout, Gc, Gc, 1.0f);
  gemm_bias_wmma<_Float16, 2><<<dim3(64, 16), blk, 0, stream>>>(lattn, wo, bo, lout, Tc, Pc, 1.0f);
}
